// NTXentLoss_17300128268344
// MI455X (gfx1250) — matrix operand order repaired and checked by reading, not compiled, not run on hardware
//
#include <hip/hip_runtime.h>
#include <hip/hip_bf16.h>
#include <math.h>

#define TEMP   0.07f
#define WEIGHT 0.5f
#define EPSN   1e-8f
#define DDIM   256

typedef _Float16 half8  __attribute__((ext_vector_type(8)));
typedef _Float16 half16 __attribute__((ext_vector_type(16)));
typedef float    float8 __attribute__((ext_vector_type(8)));

union HFrag {
    half16 v;
    struct { half8 lo; half8 hi; } p;
};

// ---------------------------------------------------------------------------
// Phase 1: L2-normalize rows to f16 copies in ws; accumulate sum_i cos(v_i,u_i)
// One wave (32 lanes) per row, 8 floats per lane.
// ---------------------------------------------------------------------------
__global__ void __launch_bounds__(256)
ntx_normalize(const float* __restrict__ img,
              const float* __restrict__ txt,
              _Float16* __restrict__ vh,
              _Float16* __restrict__ uh,
              float* __restrict__ diagsum)
{
    const int wave = threadIdx.x >> 5;
    const int lane = threadIdx.x & 31;
    const int row  = blockIdx.x * 8 + wave;

    const float4* vp = (const float4*)(img + (size_t)row * DDIM) + lane * 2;
    const float4* up = (const float4*)(txt + (size_t)row * DDIM) + lane * 2;
    float4 v0 = vp[0], v1 = vp[1];
    float4 u0 = up[0], u1 = up[1];

    float sv = v0.x*v0.x + v0.y*v0.y + v0.z*v0.z + v0.w*v0.w
             + v1.x*v1.x + v1.y*v1.y + v1.z*v1.z + v1.w*v1.w;
    float su = u0.x*u0.x + u0.y*u0.y + u0.z*u0.z + u0.w*u0.w
             + u1.x*u1.x + u1.y*u1.y + u1.z*u1.z + u1.w*u1.w;
    float sd = v0.x*u0.x + v0.y*u0.y + v0.z*u0.z + v0.w*u0.w
             + v1.x*u1.x + v1.y*u1.y + v1.z*u1.z + v1.w*u1.w;

    // wave32 butterfly reduction
    #pragma unroll
    for (int m = 1; m < 32; m <<= 1) {
        sv += __shfl_xor(sv, m);
        su += __shfl_xor(su, m);
        sd += __shfl_xor(sd, m);
    }

    const float iv = 1.0f / fmaxf(sqrtf(sv), EPSN);
    const float iu = 1.0f / fmaxf(sqrtf(su), EPSN);

    half8 hv, hu;
    hv[0] = (_Float16)(v0.x*iv); hv[1] = (_Float16)(v0.y*iv);
    hv[2] = (_Float16)(v0.z*iv); hv[3] = (_Float16)(v0.w*iv);
    hv[4] = (_Float16)(v1.x*iv); hv[5] = (_Float16)(v1.y*iv);
    hv[6] = (_Float16)(v1.z*iv); hv[7] = (_Float16)(v1.w*iv);
    hu[0] = (_Float16)(u0.x*iu); hu[1] = (_Float16)(u0.y*iu);
    hu[2] = (_Float16)(u0.z*iu); hu[3] = (_Float16)(u0.w*iu);
    hu[4] = (_Float16)(u1.x*iu); hu[5] = (_Float16)(u1.y*iu);
    hu[6] = (_Float16)(u1.z*iu); hu[7] = (_Float16)(u1.w*iu);

    *(half8*)(vh + (size_t)row * DDIM + lane * 8) = hv;
    *(half8*)(uh + (size_t)row * DDIM + lane * 8) = hu;

    if (lane == 0)
        atomicAdd(diagsum, sd * iv * iu);   // fp32-exact diagonal cosine
}

// ---------------------------------------------------------------------------
// Phase 2: fused WMMA GEMM + exp + row/col partial-sum reduction.
// Block = 4 waves (2 M x 2 N); wave tile = 64x64 = 4x4 of 16x16 WMMA tiles
// (16 f32 accumulators = 128 VGPRs -> 32 FLOP per fragment byte from L2).
// Block tile = 128(M) x 128(N). K = D = 256 in 8 steps of 32.
// ---------------------------------------------------------------------------
#define MI 4
#define NI 4

__global__ void __launch_bounds__(128)
ntx_simloss(const _Float16* __restrict__ vh,
            const _Float16* __restrict__ uh,
            float* __restrict__ rowsum,
            float* __restrict__ colsum)
{
    const int lane = threadIdx.x & 31;
    const int wave = threadIdx.x >> 5;
    const int wm   = wave >> 1;           // 0..1
    const int wn   = wave & 1;            // 0..1
    const int baseM = blockIdx.y * 128 + wm * 64;
    const int baseN = blockIdx.x * 128 + wn * 64;
    const int l16 = lane & 15;
    const int h   = lane >> 4;

    float8 acc[MI][NI] = {};

    // per-lane source rows (NT GEMM: B columns are u rows -> contiguous loads)
    const half8* arow[MI];
    const half8* brow[NI];
    #pragma unroll
    for (int mi = 0; mi < MI; ++mi)
        arow[mi] = (const half8*)(vh + (size_t)(baseM + mi * 16 + l16) * DDIM);
    #pragma unroll
    for (int ni = 0; ni < NI; ++ni)
        brow[ni] = (const half8*)(uh + (size_t)(baseN + ni * 16 + l16) * DDIM);

    #pragma unroll
    for (int k0 = 0; k0 < DDIM; k0 += 32) {
        HFrag a[MI], b[NI];
        // A 16x32 f16 layout: lane m=l16 holds K = {k0+8h..+7} and {k0+16+8h..+7}
        const int ka = (k0 >> 3) + h;
        #pragma unroll
        for (int mi = 0; mi < MI; ++mi) {
            a[mi].p.lo = arow[mi][ka];
            a[mi].p.hi = arow[mi][ka + 2];
        }
        // B 32x16 f16 layout: lane n=l16 holds K = {k0+8h..+7} and {k0+16+8h..+7}
        const int kb = (k0 >> 3) + h;
        #pragma unroll
        for (int ni = 0; ni < NI; ++ni) {
            b[ni].p.lo = brow[ni][kb];
            b[ni].p.hi = brow[ni][kb + 2];
        }

        #pragma unroll
        for (int mi = 0; mi < MI; ++mi)
            #pragma unroll
            for (int ni = 0; ni < NI; ++ni)
                acc[mi][ni] = __builtin_amdgcn_wmma_f32_16x16x32_f16(
                    false, a[mi].v, false, b[ni].v,
                    (short)0, acc[mi][ni], false, false);
    }

    // Epilogue: p = exp((s - 1)/T) (fixed shift; logits bounded by 1/T so no
    // overflow/underflow, no online-max machinery needed).
    const float invT = 1.0f / TEMP;
    float colpart[NI];
    float rowpart[MI][8];
    #pragma unroll
    for (int ni = 0; ni < NI; ++ni) colpart[ni] = 0.0f;
    #pragma unroll
    for (int mi = 0; mi < MI; ++mi)
        #pragma unroll
        for (int r = 0; r < 8; ++r) rowpart[mi][r] = 0.0f;

    #pragma unroll
    for (int mi = 0; mi < MI; ++mi)
        #pragma unroll
        for (int ni = 0; ni < NI; ++ni)
            #pragma unroll
            for (int r = 0; r < 8; ++r) {
                float p = __expf((acc[mi][ni][r] - 1.0f) * invT);
                rowpart[mi][r] += p;
                colpart[ni]    += p;
            }

    // Column sums: lane owns column baseN + ni*16 + l16 (both lane-halves add
    // their disjoint M ranges).
    #pragma unroll
    for (int ni = 0; ni < NI; ++ni)
        atomicAdd(&colsum[baseN + ni * 16 + l16], colpart[ni]);

    // Row sums: C/D layout -> row M = 8*h + r lives across the 16 lanes of a
    // half-wave in VGPR r. Butterfly over lane bits 0..3, lane l16==0 commits.
    #pragma unroll
    for (int mi = 0; mi < MI; ++mi)
        #pragma unroll
        for (int r = 0; r < 8; ++r) {
            float s = rowpart[mi][r];
            s += __shfl_xor(s, 1);
            s += __shfl_xor(s, 2);
            s += __shfl_xor(s, 4);
            s += __shfl_xor(s, 8);
            if (l16 == 0)
                atomicAdd(&rowsum[baseM + mi * 16 + 8 * h + r], s);
        }
}

// ---------------------------------------------------------------------------
// Phase 3: loss_vu = mean(log rowsum) + 1/T - mean(diag)/T ; same for colsum.
// ---------------------------------------------------------------------------
__global__ void __launch_bounds__(256)
ntx_finalize(const float* __restrict__ rowsum,
             const float* __restrict__ colsum,
             const float* __restrict__ diagsum,
             float* __restrict__ out, int N)
{
    __shared__ float sr[256];
    __shared__ float sc[256];
    float lr = 0.0f, lc = 0.0f;
    for (int i = threadIdx.x; i < N; i += 256) {
        lr += __logf(rowsum[i]);
        lc += __logf(colsum[i]);
    }
    sr[threadIdx.x] = lr;
    sc[threadIdx.x] = lc;
    __syncthreads();
    for (int s = 128; s > 0; s >>= 1) {
        if ((int)threadIdx.x < s) {
            sr[threadIdx.x] += sr[threadIdx.x + s];
            sc[threadIdx.x] += sc[threadIdx.x + s];
        }
        __syncthreads();
    }
    if (threadIdx.x == 0) {
        const float C   = 1.0f / TEMP;
        const float inv = 1.0f / (float)N;
        const float d   = diagsum[0] * inv * C;     // mean(t_ii)
        const float lvu = sr[0] * inv + C - d;
        const float luv = sc[0] * inv + C - d;
        out[0] = WEIGHT * lvu + (1.0f - WEIGHT) * luv;
        out[1] = lvu;
        out[2] = luv;
    }
}

// ---------------------------------------------------------------------------
extern "C" void kernel_launch(void* const* d_in, const int* in_sizes, int n_in,
                              void* d_out, int out_size, void* d_ws, size_t ws_size,
                              hipStream_t stream)
{
    (void)n_in; (void)out_size; (void)ws_size;
    const float* img = (const float*)d_in[0];
    const float* txt = (const float*)d_in[1];
    const int N = in_sizes[0] / DDIM;           // 16384

    char* ws = (char*)d_ws;
    const size_t featBytes = (size_t)N * DDIM * sizeof(_Float16);  // 8 MB each
    _Float16* vh     = (_Float16*)ws;
    _Float16* uh     = (_Float16*)(ws + featBytes);
    float*    rowsum = (float*)(ws + 2 * featBytes);
    float*    colsum = rowsum + N;
    float*    diagsum = colsum + N;

    // zero accumulators every call (ws is poisoned; atomics accumulate)
    hipMemsetAsync(rowsum, 0, (2 * (size_t)N + 1) * sizeof(float), stream);

    ntx_normalize<<<N / 8, 256, 0, stream>>>(img, txt, vh, uh, diagsum);

    dim3 grid(N / 128, N / 128);                // 128 x 128 blocks
    ntx_simloss<<<grid, 128, 0, stream>>>(vh, uh, rowsum, colsum);

    ntx_finalize<<<1, 256, 0, stream>>>(rowsum, colsum, diagsum, (float*)d_out, N);
}
